// GradientAdaptiveEdgeModule_11038065951511
// MI455X (gfx1250) — matrix operand order repaired and checked by reading, not compiled, not run on hardware
//
#include <hip/hip_runtime.h>
#include <hip/hip_bf16.h>
#include <math.h>

typedef __bf16 bf16_t;
typedef __attribute__((ext_vector_type(16))) __bf16 v16bf;
typedef __attribute__((ext_vector_type(8)))  __bf16 v8bf;
typedef __attribute__((ext_vector_type(8)))  float  v8f;

#define HW   16384      // 128*128
#define NCH  64
#define NPIX 32768      // B*HW
#define EPSV 1e-6f

// ---- WMMA A-fragment K map (contiguous order, matches B, wave32) -----
// lane<16 -> M=lane,    elems 0..15 = K0..15
// lane>=16 -> M=lane-16, elems 0..15 = K16..31
__device__ __forceinline__ int amapK(int lane, int e) {
    int kb = (lane < 16) ? 0 : 16;
    return kb + e;
}
// B (32x16): N = lane%16; lane<16 holds rows K=0..15, lane>=16 K=16..31.
// With NHWC activations those 16 K's are 16 consecutive bf16 -> one v16bf load.

// ---------------------------------------------------------------------------
__global__ void k_zero_f32(float* __restrict__ p, int n) {
    int i = blockIdx.x * 256 + threadIdx.x;
    if (i < n) p[i] = 0.0f;
}

// pack ga_w1 [64,128] -> A fragments [mt(4)][ks(4)][lane(32)][e(16)]
__global__ void k_pack_w1(const float* __restrict__ w, bf16_t* __restrict__ d) {
    int i = blockIdx.x * 256 + threadIdx.x;          // 8192
    int e = i & 15, lane = (i >> 4) & 31, rest = i >> 9;
    int ks = rest & 3, mt = rest >> 2;
    int crow = mt * 16 + (lane & 15);
    d[i] = (bf16_t)w[crow * 128 + ks * 32 + amapK(lane, e)];
}

// pack ga_w2 [16,64] -> [ks(2)][lane(32)][e(16)]
__global__ void k_pack_w2(const float* __restrict__ w, bf16_t* __restrict__ d) {
    int i = blockIdx.x * 256 + threadIdx.x;          // 1024
    int e = i & 15, lane = (i >> 4) & 31, ks = i >> 9;
    d[i] = (bf16_t)w[(lane & 15) * 64 + ks * 32 + amapK(lane, e)];
}

// pack ec_w [64,64,3,3] -> [mt(4)][s(18)][lane(32)][e(16)], s = (tap, 32-chunk)
__global__ void k_pack_wc(const float* __restrict__ w, bf16_t* __restrict__ d) {
    int i = blockIdx.x * 256 + threadIdx.x;          // 36864
    int e = i & 15, lane = (i >> 4) & 31, rest = i >> 9;
    int s = rest % 18, mt = rest / 18;
    int tap = s >> 1, cb = (s & 1) << 5;
    int crow = mt * 16 + (lane & 15);
    d[i] = (bf16_t)w[((size_t)crow * 64 + cb + amapK(lane, e)) * 9 + tap];
}

// Sobel depthwise grads; writes NCHW mag/dir (for spatial windows) and
// channel-last bf16 activations: gf[q*128 + c] = gx, gf[q*128+64+c] = gy,
// xb[q*64 + c] = x  (q = global pixel)
__global__ void k_sobel_prep(const float* __restrict__ x,
                             const float* __restrict__ wgx,
                             const float* __restrict__ wgy,
                             float* __restrict__ mag, float* __restrict__ dir,
                             bf16_t* __restrict__ gf, bf16_t* __restrict__ xb) {
    int i = blockIdx.x * 256 + threadIdx.x;          // over B*C*HW (NCHW)
    int p = i & (HW - 1);
    int hh = p >> 7, ww = p & 127;
    int c = (i >> 14) & 63;
    float sx = 0.f, sy = 0.f;
    #pragma unroll
    for (int tap = 0; tap < 9; ++tap) {
        int dy = tap / 3 - 1, dx = tap % 3 - 1;
        int yy = hh + dy, xx = ww + dx;
        if ((unsigned)yy < 128u && (unsigned)xx < 128u) {
            float v = x[i + dy * 128 + dx];
            sx += wgx[c * 9 + tap] * v;
            sy += wgy[c * 9 + tap] * v;
        }
    }
    mag[i] = sqrtf(sx * sx + sy * sy + EPSV);
    dir[i] = atan2f(sy, sx);
    size_t q = (size_t)(i >> 20) * HW + p;           // global pixel
    gf[q * 128 + c]      = (bf16_t)sx;
    gf[q * 128 + 64 + c] = (bf16_t)sy;
    xb[q * 64 + c]       = (bf16_t)x[i];
}

// 5x5 weighted circular-stat consistency, channel-mean via atomicAdd
__global__ void k_consistency(const float* __restrict__ mag,
                              const float* __restrict__ dir,
                              float* __restrict__ cons_sum) {
    int i = blockIdx.x * 256 + threadIdx.x;          // over B*C*HW
    int p = i & (HW - 1);
    int hh = p >> 7, ww = p & 127;
    float dw[25], mw[25];
    float sm = 0.f, sdm = 0.f;
    int n = 0;
    #pragma unroll
    for (int dy = -2; dy <= 2; ++dy) {
        #pragma unroll
        for (int dx = -2; dx <= 2; ++dx) {
            int yy = hh + dy, xx = ww + dx;
            bool v = (unsigned)yy < 128u && (unsigned)xx < 128u;
            float m = v ? mag[i + dy * 128 + dx] : 0.f;
            float d = v ? dir[i + dy * 128 + dx] : 0.f;
            mw[n] = m; dw[n] = d;
            sm += m; sdm += d * m; ++n;
        }
    }
    float wmean = sdm / (sm + EPSV);
    float sv = 0.f;
    #pragma unroll
    for (int j = 0; j < 25; ++j) {
        float t = dw[j] - wmean;
        sv += t * t * mw[j];
    }
    float cons = 1.f - tanhf(sqrtf(sv / (sm + EPSV)));
    atomicAdd(&cons_sum[(i >> 20) * HW + p], cons);
}

// ga1 = relu(W1[64x128]*gf + b1): WMMA GEMM, NHWC in/out, all-vector loads
__global__ void k_ga1_wmma(const bf16_t* __restrict__ gf,
                           const bf16_t* __restrict__ wpack,
                           const float* __restrict__ b1,
                           bf16_t* __restrict__ out) {
    int wid  = (blockIdx.x * blockDim.x + threadIdx.x) >> 5;
    int lane = threadIdx.x & 31;
    const int PT = NPIX / 16;
    int mt = wid / PT;
    int pt = wid % PT;
    size_t q = (size_t)pt * 16 + (lane & 15);        // this lane's pixel
    const v16bf* ap = (const v16bf*)wpack;
    int kb16 = (lane < 16) ? 0 : 16;
    v8f acc = {};
    #pragma unroll
    for (int ks = 0; ks < 4; ++ks) {
        v16bf a  = ap[(mt * 4 + ks) * 32 + lane];
        v16bf bm = *(const v16bf*)(gf + q * 128 + ks * 32 + kb16);
        acc = __builtin_amdgcn_wmma_f32_16x16x32_bf16(false, a, false, bm,
                                                      (short)0, acc, false, false);
    }
    int cb8 = mt * 16 + ((lane < 16) ? 0 : 8);
    v8bf st;
    #pragma unroll
    for (int i = 0; i < 8; ++i) {
        float v = acc[i] + b1[cb8 + i];
        st[i] = (bf16_t)(v > 0.f ? v : 0.f);
    }
    *(v8bf*)(out + q * 64 + cb8) = st;
}

// ga2 = relu(W2[16x64]*ga1 + b2): WMMA GEMM, NHWC f32 out (16ch)
__global__ void k_ga2_wmma(const bf16_t* __restrict__ in,
                           const bf16_t* __restrict__ wpack,
                           const float* __restrict__ b2,
                           float* __restrict__ out) {
    int wid  = (blockIdx.x * blockDim.x + threadIdx.x) >> 5;
    int lane = threadIdx.x & 31;
    size_t q = (size_t)wid * 16 + (lane & 15);
    const v16bf* ap = (const v16bf*)wpack;
    int kb16 = (lane < 16) ? 0 : 16;
    v8f acc = {};
    #pragma unroll
    for (int ks = 0; ks < 2; ++ks) {
        v16bf a  = ap[ks * 32 + lane];
        v16bf bm = *(const v16bf*)(in + q * 64 + ks * 32 + kb16);
        acc = __builtin_amdgcn_wmma_f32_16x16x32_bf16(false, a, false, bm,
                                                      (short)0, acc, false, false);
    }
    int cb8 = (lane < 16) ? 0 : 8;
    v8f st;
    #pragma unroll
    for (int i = 0; i < 8; ++i) {
        float v = acc[i] + b2[cb8 + i];
        st[i] = v > 0.f ? v : 0.f;
    }
    *(v8f*)(out + q * 16 + cb8) = st;
}

// cd branch: conv3x3 16->16 relu, 1x1 16->1 sigmoid, * channel-mean consistency
__global__ void k_cd_ew(const float* __restrict__ ga2,   // NHWC [q][16]
                        const float* __restrict__ w1, const float* __restrict__ bias1,
                        const float* __restrict__ w2, const float* __restrict__ b2,
                        const float* __restrict__ cons_sum, float* __restrict__ ew) {
    int idx = blockIdx.x * 256 + threadIdx.x;        // over B*HW
    int b = idx >> 14;
    int p = idx & (HW - 1);
    int hh = p >> 7, ww = p & 127;
    float s1[16];
    #pragma unroll
    for (int oc = 0; oc < 16; ++oc) s1[oc] = bias1[oc];
    #pragma unroll
    for (int tap = 0; tap < 9; ++tap) {
        int dy = tap / 3 - 1, dx = tap % 3 - 1;
        int yy = hh + dy, xx = ww + dx;
        if ((unsigned)yy < 128u && (unsigned)xx < 128u) {
            const float* g = ga2 + ((size_t)b * HW + yy * 128 + xx) * 16;
            #pragma unroll
            for (int ic = 0; ic < 16; ++ic) {
                float gv = g[ic];
                #pragma unroll
                for (int oc = 0; oc < 16; ++oc)
                    s1[oc] += w1[(oc * 16 + ic) * 9 + tap] * gv;
            }
        }
    }
    float z = b2[0];
    #pragma unroll
    for (int oc = 0; oc < 16; ++oc) {
        float v = s1[oc];
        z += w2[oc] * (v > 0.f ? v : 0.f);
    }
    float sig = 1.f / (1.f + expf(-z));
    ew[idx] = sig * cons_sum[idx] * (1.0f / 64.0f);
}

// dense 3x3 conv 64->64 implicit WMMA GEMM, NHWC in; 18 K-steps of 32.
// fuse_final: out = x + alpha*ew*(conv+bias) -> NCHW f32 d_out; else relu->NHWC bf16
__global__ void k_conv3x3_wmma(const bf16_t* __restrict__ in,
                               const bf16_t* __restrict__ wpack,
                               const float* __restrict__ bias,
                               bf16_t* __restrict__ out_bf,
                               const float* __restrict__ xres,
                               const float* __restrict__ ew,
                               const float* __restrict__ alpha_p,
                               float* __restrict__ out_f,
                               int fuse_final) {
    int wid  = (blockIdx.x * blockDim.x + threadIdx.x) >> 5;
    int lane = threadIdx.x & 31;
    const int PT = NPIX / 16;
    int mt = wid / PT;
    int pt = wid % PT;
    int q0 = pt * 16;
    int b  = q0 >> 14;
    int p0 = q0 & (HW - 1);
    int h  = p0 >> 7;
    int xcol = (p0 & 127) + (lane & 15);
    const v16bf* ap = (const v16bf*)wpack;
    int kb16 = (lane < 16) ? 0 : 16;
    v16bf zz = {};
    v8f acc = {};
    for (int s = 0; s < 18; ++s) {
        int tap = s >> 1;
        int cb  = (s & 1) << 5;
        int dy = tap / 3 - 1, dx = tap % 3 - 1;
        v16bf a = ap[(mt * 18 + s) * 32 + lane];
        int yy = h + dy, xx = xcol + dx;
        bool valid = (unsigned)yy < 128u && (unsigned)xx < 128u;
        v16bf bm = valid
            ? *(const v16bf*)(in + ((size_t)b * HW + yy * 128 + xx) * 64 + cb + kb16)
            : zz;
        acc = __builtin_amdgcn_wmma_f32_16x16x32_bf16(false, a, false, bm,
                                                      (short)0, acc, false, false);
    }
    int p = h * 128 + xcol;
    int cb8 = mt * 16 + ((lane < 16) ? 0 : 8);
    if (fuse_final) {
        float aw = alpha_p[0] * ew[(size_t)b * HW + p];
        #pragma unroll
        for (int i = 0; i < 8; ++i) {
            size_t idx = ((size_t)b * NCH + cb8 + i) * HW + p;   // NCHW output
            out_f[idx] = xres[idx] + aw * (acc[i] + bias[cb8 + i]);
        }
    } else {
        v8bf st;
        #pragma unroll
        for (int i = 0; i < 8; ++i) {
            float v = acc[i] + bias[cb8 + i];
            st[i] = (bf16_t)(v > 0.f ? v : 0.f);
        }
        *(v8bf*)(out_bf + ((size_t)b * HW + p) * 64 + cb8) = st;
    }
}

// ---------------------------------------------------------------------------
extern "C" void kernel_launch(void* const* d_in, const int* in_sizes, int n_in,
                              void* d_out, int out_size, void* d_ws, size_t ws_size,
                              hipStream_t stream) {
    const float* x     = (const float*)d_in[0];
    const float* w_gx  = (const float*)d_in[1];
    const float* w_gy  = (const float*)d_in[2];
    const float* ga_w1 = (const float*)d_in[3];
    const float* ga_b1 = (const float*)d_in[4];
    const float* ga_w2 = (const float*)d_in[5];
    const float* ga_b2 = (const float*)d_in[6];
    const float* cd_w1 = (const float*)d_in[7];
    const float* cd_b1 = (const float*)d_in[8];
    const float* cd_w2 = (const float*)d_in[9];
    const float* cd_b2 = (const float*)d_in[10];
    const float* ec_w1 = (const float*)d_in[11];
    const float* ec_b1 = (const float*)d_in[12];
    const float* ec_w2 = (const float*)d_in[13];
    const float* ec_b2 = (const float*)d_in[14];
    const float* alpha = (const float*)d_in[15];
    float* out = (float*)d_out;

    const size_t BCHW = (size_t)2 * NCH * HW;        // 2097152
    size_t off = 0;
    auto carve = [&](size_t bytes) -> char* {
        char* p = (char*)d_ws + off;
        off += (bytes + 255) & ~(size_t)255;
        return p;
    };
    float*  mag      = (float*) carve(BCHW * 4);
    float*  dir      = (float*) carve(BCHW * 4);
    bf16_t* gf_bf    = (bf16_t*)carve((size_t)NPIX * 128 * 2);   // NHWC 128ch
    bf16_t* x_bf     = (bf16_t*)carve(BCHW * 2);                 // NHWC 64ch
    bf16_t* ga1_bf   = (bf16_t*)carve(BCHW * 2);                 // NHWC 64ch
    float*  ga2_f    = (float*) carve((size_t)NPIX * 16 * 4);    // NHWC 16ch
    bf16_t* ee1_bf   = (bf16_t*)carve(BCHW * 2);                 // NHWC 64ch
    float*  cons_sum = (float*) carve((size_t)NPIX * 4);
    float*  ew       = (float*) carve((size_t)NPIX * 4);
    bf16_t* wga1_pk  = (bf16_t*)carve((size_t)8192 * 2);
    bf16_t* wga2_pk  = (bf16_t*)carve((size_t)1024 * 2);
    bf16_t* wec1_pk  = (bf16_t*)carve((size_t)36864 * 2);
    bf16_t* wec2_pk  = (bf16_t*)carve((size_t)36864 * 2);
    (void)ws_size; (void)n_in; (void)in_sizes; (void)out_size;

    // weight fragment packing (bf16, WMMA A-layout)
    k_pack_w1<<<32, 256, 0, stream>>>(ga_w1, wga1_pk);
    k_pack_w2<<<4, 256, 0, stream>>>(ga_w2, wga2_pk);
    k_pack_wc<<<144, 256, 0, stream>>>(ec_w1, wec1_pk);
    k_pack_wc<<<144, 256, 0, stream>>>(ec_w2, wec2_pk);

    // sobel grads + mag/dir + NHWC bf16 activations
    k_sobel_prep<<<(int)(BCHW / 256), 256, 0, stream>>>(x, w_gx, w_gy, mag, dir,
                                                        gf_bf, x_bf);
    // consistency channel-mean accumulator
    k_zero_f32<<<NPIX / 256, 256, 0, stream>>>(cons_sum, NPIX);
    k_consistency<<<(int)(BCHW / 256), 256, 0, stream>>>(mag, dir, cons_sum);

    // gradient analyzer GEMMs (WMMA, vector loads only)
    k_ga1_wmma<<<1024, 256, 0, stream>>>(gf_bf, wga1_pk, ga_b1, ga1_bf);
    k_ga2_wmma<<<256, 256, 0, stream>>>(ga1_bf, wga2_pk, ga_b2, ga2_f);

    // consistency detector -> scalar edge weight map
    k_cd_ew<<<NPIX / 256, 256, 0, stream>>>(ga2_f, cd_w1, cd_b1, cd_w2, cd_b2,
                                            cons_sum, ew);

    // edge branch: two dense 3x3 convs as implicit WMMA GEMMs; second fused
    // with the residual combine, writing NCHW f32 d_out directly
    k_conv3x3_wmma<<<1024, 256, 0, stream>>>(x_bf, wec1_pk, ec_b1, ee1_bf,
                                             nullptr, nullptr, nullptr, nullptr, 0);
    k_conv3x3_wmma<<<1024, 256, 0, stream>>>(ee1_bf, wec2_pk, ec_b2, nullptr,
                                             x, ew, alpha, out, 1);
}
